// Vanilla_9878424781199
// MI455X (gfx1250) — matrix operand order repaired and checked by reading, not compiled, not run on hardware
//
#include <hip/hip_runtime.h>
#include <hip/hip_bf16.h>

// ---------------------------------------------------------------------------
// Problem constants (fixed by the reference: M=512, N=4096, D=128, H=32,
// L=4096, P=3584; note P + M == L).
// ---------------------------------------------------------------------------
#define M_ 512
#define N_ 4096
#define D_ 128
#define H_ 32
#define L_ 4096
#define P_ 3584

typedef __attribute__((ext_vector_type(16))) __bf16   v16bf;
typedef __attribute__((ext_vector_type(8)))  __bf16   v8bf;
typedef __attribute__((ext_vector_type(8)))  float    v8f;
typedef __attribute__((ext_vector_type(4)))  float    v4f;
typedef __attribute__((ext_vector_type(8)))  unsigned v8u;
typedef __attribute__((ext_vector_type(4)))  unsigned v4u;
typedef __attribute__((ext_vector_type(2)))  unsigned v2u;
typedef __attribute__((ext_vector_type(8)))  int      v8i;
typedef __attribute__((ext_vector_type(4)))  int      v4i;

// Pack two f32 into two bf16 (round-to-nearest, ties away from zero).
static __device__ __forceinline__ unsigned pack_bf16_2(float f0, float f1) {
    unsigned u0 = __float_as_uint(f0) + 0x8000u;
    unsigned u1 = __float_as_uint(f1) + 0x8000u;
#if __has_builtin(__builtin_amdgcn_perm)
    return __builtin_amdgcn_perm(u1, u0, 0x07060302u);
#else
    return (u0 >> 16) | (u1 & 0xFFFF0000u);
#endif
}

// scalar f32 -> bf16 RNE (low-frequency paths)
static __device__ __forceinline__ __bf16 f2bf(float x) {
    unsigned u = __float_as_uint(x);
    u += 0x7FFFu + ((u >> 16) & 1u);
    unsigned short h = (unsigned short)(u >> 16);
    return __builtin_bit_cast(__bf16, h);
}

static __device__ __forceinline__ v16bf join8(v8bf lo, v8bf hi) {
    v16bf r;
#pragma unroll
    for (int i = 0; i < 8; ++i) { r[i] = lo[i]; r[i + 8] = hi[i]; }
    return r;
}

static __device__ __forceinline__ v8f wmma_bf16(v16bf a, v16bf b, v8f c) {
    return __builtin_amdgcn_wmma_f32_16x16x32_bf16(
        false, a, false, b, (short)0, c, false, false);
}

// ---------------------------------------------------------------------------
// Kernel 0: convert X (f32, [M][N]) -> Xb (bf16). 8 elements per thread.
// ---------------------------------------------------------------------------
__global__ __launch_bounds__(256)
void cvt_x(const float* __restrict__ X, __bf16* __restrict__ Xb)
{
    const int i = blockIdx.x * blockDim.x + threadIdx.x;
    const float* p = X + (size_t)i * 8;
    v4f a = *(const v4f*)p;
    v4f b = *(const v4f*)(p + 4);
    v4u w;
    w[0] = pack_bf16_2(a[0], a[1]);
    w[1] = pack_bf16_2(a[2], a[3]);
    w[2] = pack_bf16_2(b[0], b[1]);
    w[3] = pack_bf16_2(b[2], b[3]);
    *((v4u*)Xb + i) = w;
}

// ---------------------------------------------------------------------------
// Kernel 1: y = X @ W^T for W in {Wq, Wk, Wv}. Block = 128 thr (4 waves),
// tile 128x128; wave = 64x64 (4x4 WMMA accumulators, 16 WMMA per k-step).
// ---------------------------------------------------------------------------
__global__ __launch_bounds__(128)
void qkv_gemm(const __bf16* __restrict__ Xb, const float* __restrict__ Wq,
              const float* __restrict__ Wk, const float* __restrict__ Wv,
              __bf16* __restrict__ Qb, __bf16* __restrict__ Kb,
              __bf16* __restrict__ Vb)
{
    const int lane = threadIdx.x & 31;
    const int wave = threadIdx.x >> 5;
    const int lm   = lane & 15;
    const int lh   = lane >> 4;

    const int nblk = blockIdx.x * 128;
    const int mblk = blockIdx.y * 128;
    const int sel  = blockIdx.z;
    const float* __restrict__ W = (sel == 0) ? Wq : (sel == 1) ? Wk : Wv;

    const int mbase = mblk + (wave >> 1) * 64;
    const int nbase = nblk + (wave & 1) * 64;

    v8f acc[4][4];
#pragma unroll
    for (int tm = 0; tm < 4; ++tm)
#pragma unroll
        for (int tn = 0; tn < 4; ++tn)
#pragma unroll
            for (int r = 0; r < 8; ++r) acc[tm][tn][r] = 0.0f;

    for (int k0 = 0; k0 < N_; k0 += 32) {
        v16bf a[4];
#pragma unroll
        for (int tm = 0; tm < 4; ++tm) {
            const __bf16* p = Xb + (size_t)(mbase + tm * 16 + lm) * N_ + k0 + lh * 16;
            a[tm] = join8(*(const v8bf*)p, *(const v8bf*)(p + 8));
        }
#pragma unroll
        for (int tn = 0; tn < 4; ++tn) {
            const float* p = W + (size_t)(nbase + tn * 16 + lm) * N_ + k0 + lh * 16;
            v4f x0 = *(const v4f*)(p);
            v4f x1 = *(const v4f*)(p + 4);
            v4f x2 = *(const v4f*)(p + 8);
            v4f x3 = *(const v4f*)(p + 12);
            v8u w;
            w[0] = pack_bf16_2(x0[0], x0[1]);
            w[1] = pack_bf16_2(x0[2], x0[3]);
            w[2] = pack_bf16_2(x1[0], x1[1]);
            w[3] = pack_bf16_2(x1[2], x1[3]);
            w[4] = pack_bf16_2(x2[0], x2[1]);
            w[5] = pack_bf16_2(x2[2], x2[3]);
            w[6] = pack_bf16_2(x3[0], x3[1]);
            w[7] = pack_bf16_2(x3[2], x3[3]);
            const v16bf b = __builtin_bit_cast(v16bf, w);
#pragma unroll
            for (int tm = 0; tm < 4; ++tm)
                acc[tm][tn] = wmma_bf16(a[tm], b, acc[tm][tn]);
        }
    }

#pragma unroll
    for (int tm = 0; tm < 4; ++tm)
#pragma unroll
        for (int tn = 0; tn < 4; ++tn)
#pragma unroll
            for (int r = 0; r < 8; ++r) {
                const int m = mbase + tm * 16 + lh * 8 + r;
                const int n = nbase + tn * 16 + lm;
                const int h = n >> 7;
                const int d = n & 127;
                const __bf16 val = f2bf(acc[tm][tn][r]);
                if (sel == 0)
                    Qb[((size_t)h * M_ + m) * D_ + d] = val;
                else if (sel == 1)
                    Kb[((size_t)h * L_ + P_ + m) * D_ + d] = val;
                else
                    Vb[((size_t)h * L_ + P_ + m) * D_ + d] = val;
            }
}

// ---------------------------------------------------------------------------
// Kernel 2: convert input f32 caches rows [0,P) to bf16 in Kb/Vb.
// ---------------------------------------------------------------------------
__global__ __launch_bounds__(256)
void cvt_cache(const float* __restrict__ cK, const float* __restrict__ cV,
               __bf16* __restrict__ Kb, __bf16* __restrict__ Vb)
{
    const int nvec = H_ * P_ * (D_ / 4);
    int i = blockIdx.x * blockDim.x + threadIdx.x;
    if (i >= 2 * nvec) return;
    const float* __restrict__ src;
    __bf16* __restrict__ dst;
    if (i >= nvec) { i -= nvec; src = cV; dst = Vb; }
    else           { src = cK; dst = Kb; }
    const int d4   = i & 31;
    const int rest = i >> 5;
    const int l    = rest % P_;
    const int h    = rest / P_;
    const size_t fidx = ((size_t)(h * L_ + l)) * (D_ / 4) + d4;
    v4f v = *((const v4f*)src + fidx);
    v2u o;
    o[0] = pack_bf16_2(v[0], v[1]);
    o[1] = pack_bf16_2(v[2], v[3]);
    *((v2u*)dst + fidx) = o;
}

// ---------------------------------------------------------------------------
// Kernel 3: flash attention. K chunks are staged into LDS by the Tensor Data
// Mover (tensor_load_to_lds), with D#-generated padding producing the
// bank-conflict-free [32][144] layout; completion via s_wait_tensorcnt +
// workgroup barrier. V is staged transposed manually (TDM has no transpose).
// ---------------------------------------------------------------------------
#define ATTN_USE_TDM (__has_builtin(__builtin_amdgcn_tensor_load_to_lds))

__global__ __launch_bounds__(128)
void attn(const __bf16* __restrict__ Qb, const __bf16* __restrict__ Kb,
          const __bf16* __restrict__ Vb, float* __restrict__ out)
{
    __shared__ __bf16 Klds[32][144];      // keys x d (row = 256B data + 32B pad)
    __shared__ __bf16 Vt[128][48];        // d x keys (pad 32->48, 96B rows)
    __shared__ __bf16 Plds[4][16][48];    // per-wave P tile, rows x keys

    const int tid  = threadIdx.x;
    const int lane = tid & 31;
    const int wave = tid >> 5;
    const int lm   = lane & 15;
    const int lh   = lane >> 4;

    const int h  = blockIdx.y;
    const int q0 = blockIdx.x * 64 + wave * 16;

    const __bf16* __restrict__ Qh = Qb + (size_t)h * M_ * D_;
    const __bf16* __restrict__ Kh = Kb + (size_t)h * L_ * D_;
    const __bf16* __restrict__ Vh = Vb + (size_t)h * L_ * D_;

#if ATTN_USE_TDM
    // --- constant parts of the K-tile tensor descriptor (D#) ---
    // group0: [1:0] count=1; [63:32] lds_addr; [120:64] global_addr;
    //         [127:126] type=2
    // group1: [17:16] data_size=1 (2B); [20] pad_enable;
    //         [24:22] pad_interval=5 (64 DW = one 256B row);
    //         [31:25] pad_amount=7 (8 DW = 32B)  -> row stride 288B == [144]
    //         tensor_dim0=tile_dim0=128, tensor_dim1=tile_dim1=32, stride0=128
    const unsigned ldsK = (unsigned)(uintptr_t)&Klds[0][0];
    const unsigned g1w0 = (1u << 16) | (1u << 20) | (5u << 22) | (7u << 25);
    const unsigned td0 = 128u, td1 = 32u, tile0 = 128u, tile1 = 32u, str0 = 128u;
    v8i g1;
    g1[0] = (int)g1w0;
    g1[1] = (int)((td0 & 0xFFFFu) << 16);            // barrier addr 0 | td0.lo
    g1[2] = (int)((td0 >> 16) | ((td1 & 0xFFFFu) << 16));
    g1[3] = (int)((td1 >> 16) | (tile0 << 16));
    g1[4] = (int)(tile1 | (0u << 16));               // tile_dim2 = 0
    g1[5] = (int)str0;                               // stride0 lo32
    g1[6] = 0;                                       // stride0 hi | stride1 lo
    g1[7] = 0;                                       // stride1 hi
    const v4i gz4 = {0, 0, 0, 0};
    const v8i gz8 = {0, 0, 0, 0, 0, 0, 0, 0};
#endif

    // Preload Q A-fragments: 4 chunks of k=32 covering D=128
    v16bf qf[4];
#pragma unroll
    for (int j = 0; j < 4; ++j) {
        const __bf16* p = Qh + (size_t)(q0 + lm) * D_ + j * 32 + lh * 16;
        qf[j] = join8(*(const v8bf*)p, *(const v8bf*)(p + 8));
    }

    v8f o[8];
#pragma unroll
    for (int t = 0; t < 8; ++t)
#pragma unroll
        for (int r = 0; r < 8; ++r) o[t][r] = 0.0f;
    float mrun[8], lrun[8];
#pragma unroll
    for (int r = 0; r < 8; ++r) { mrun[r] = -3.0e38f; lrun[r] = 0.0f; }

    for (int kb = 0; kb < L_; kb += 32) {
#if ATTN_USE_TDM
        // ---- K chunk via Tensor Data Mover (one instruction, wave 0) ----
        if (wave == 0) {
            const unsigned long long ga =
                (unsigned long long)(uintptr_t)(Kh + (size_t)kb * D_);
            v4u g0;
            g0[0] = 1u;                                   // count=1
            g0[1] = ldsK;                                 // lds_addr
            g0[2] = (unsigned)(ga & 0xFFFFFFFFu);         // global_addr lo
            g0[3] = (unsigned)((ga >> 32) & 0x01FFFFFFu) | 0x80000000u; // hi|type=2
            __builtin_amdgcn_tensor_load_to_lds(g0, g1, gz4, gz4, gz8, 0);
        }
#endif
        // ---- V chunk staged transposed (and K, when no TDM) ----
        {
            const int row = tid >> 2;          // 0..31 key within chunk
            const int seg = (tid & 3) * 32;    // d segment
#if !ATTN_USE_TDM
            const __bf16* ps = Kh + (size_t)(kb + row) * D_ + seg;
            *(v16bf*)&Klds[row][seg]      = *(const v16bf*)ps;
            *(v16bf*)&Klds[row][seg + 16] = *(const v16bf*)(ps + 16);
#endif
            const __bf16* pv = Vh + (size_t)(kb + row) * D_ + seg;
#pragma unroll
            for (int e = 0; e < 32; ++e) Vt[seg + e][row] = pv[e];
        }
#if ATTN_USE_TDM
        if (wave == 0) __builtin_amdgcn_s_wait_tensorcnt(0);
#endif
        __syncthreads();

        // ---- scores: 16 rows x 32 keys = 2 C-fragments, 4 k-steps each ----
        v8f s[2];
#pragma unroll
        for (int t = 0; t < 2; ++t) {
#pragma unroll
            for (int r = 0; r < 8; ++r) s[t][r] = 0.0f;
#pragma unroll
            for (int j = 0; j < 4; ++j) {
                v16bf bk = *(const v16bf*)&Klds[t * 16 + lm][j * 32 + lh * 16];
                s[t] = wmma_bf16(qf[j], bk, s[t]);
            }
        }

        // ---- online softmax update ----
        float corr[8];
#pragma unroll
        for (int r = 0; r < 8; ++r) {
            float cm = fmaxf(s[0][r], s[1][r]);
#pragma unroll
            for (int sh = 1; sh < 16; sh <<= 1) cm = fmaxf(cm, __shfl_xor(cm, sh));
            const float mnew = fmaxf(mrun[r], cm);
            corr[r] = __expf(mrun[r] - mnew);
            const float p0 = __expf(s[0][r] - mnew);
            const float p1 = __expf(s[1][r] - mnew);
            const int rowr = lh * 8 + r;
            Plds[wave][rowr][lm]      = f2bf(p0);
            Plds[wave][rowr][16 + lm] = f2bf(p1);
            float ps = p0 + p1;
#pragma unroll
            for (int sh = 1; sh < 16; sh <<= 1) ps += __shfl_xor(ps, sh);
            lrun[r] = lrun[r] * corr[r] + ps;
            mrun[r] = mnew;
        }
#pragma unroll
        for (int t = 0; t < 8; ++t)
#pragma unroll
            for (int r = 0; r < 8; ++r) o[t][r] *= corr[r];

        // ---- reshape P (C-layout -> A-layout) via per-wave LDS ----
        asm volatile("" ::: "memory");
        const __bf16* pp0 = &Plds[wave][lm][lh * 16];
        const __bf16* pp1 = &Plds[wave][lm][lh * 16 + 8];
        v16bf pa = join8(*(const v8bf*)pp0, *(const v8bf*)pp1);

        // ---- O += P @ V over 8 d-tiles ----
#pragma unroll
        for (int t = 0; t < 8; ++t) {
            v16bf bv = *(const v16bf*)&Vt[t * 16 + lm][lh * 16];
            o[t] = wmma_bf16(pa, bv, o[t]);
        }
        __syncthreads();
    }

    // ---- normalize and store ----
    float rinv[8];
#pragma unroll
    for (int r = 0; r < 8; ++r) rinv[r] = 1.0f / lrun[r];
#pragma unroll
    for (int t = 0; t < 8; ++t)
#pragma unroll
        for (int r = 0; r < 8; ++r) {
            const int mrow = q0 + lh * 8 + r;
            const int d    = t * 16 + lm;
            out[(size_t)mrow * N_ + h * D_ + d] = o[t][r] * rinv[r];
        }
}

// ---------------------------------------------------------------------------
// Launch. Workspace (bf16): Qb 4MB, Kb 32MB, Vb 32MB, Xb 4MB  -> ~72 MB.
// ---------------------------------------------------------------------------
extern "C" void kernel_launch(void* const* d_in, const int* in_sizes, int n_in,
                              void* d_out, int out_size, void* d_ws, size_t ws_size,
                              hipStream_t stream)
{
    (void)in_sizes; (void)n_in; (void)out_size; (void)ws_size;
    const float* X  = (const float*)d_in[0];
    const float* Wq = (const float*)d_in[1];
    const float* Wk = (const float*)d_in[2];
    const float* Wv = (const float*)d_in[3];
    const float* cK = (const float*)d_in[4];
    const float* cV = (const float*)d_in[5];
    float* out = (float*)d_out;

    __bf16* Qb = (__bf16*)d_ws;
    __bf16* Kb = Qb + (size_t)H_ * M_ * D_;
    __bf16* Vb = Kb + (size_t)H_ * L_ * D_;
    __bf16* Xb = Vb + (size_t)H_ * L_ * D_;

    cvt_x<<<(M_ * N_ / 8) / 256, 256, 0, stream>>>(X, Xb);

    const int total = 2 * H_ * P_ * (D_ / 4);
    cvt_cache<<<(total + 255) / 256, 256, 0, stream>>>(cK, cV, Kb, Vb);

    qkv_gemm<<<dim3(N_ / 128, M_ / 128, 3), 128, 0, stream>>>(Xb, Wq, Wk, Wv, Qb, Kb, Vb);

    attn<<<dim3(M_ / 64, H_), 128, 0, stream>>>(Qb, Kb, Vb, out);
}
